// RNN_39797166965205
// MI455X (gfx1250) — matrix operand order repaired and checked by reading, not compiled, not run on hardware
//
#include <hip/hip_runtime.h>
#include <hip/hip_bf16.h>

typedef __attribute__((ext_vector_type(16))) __bf16 v16bf;
typedef __attribute__((ext_vector_type(8)))  __bf16 v8bf;
typedef __attribute__((ext_vector_type(8)))  float  v8f;

constexpr int BB   = 4096;
constexpr int TT   = 36;
constexpr int FF   = 128;
constexpr int HH   = 256;
constexpr int D1   = 64;
constexpr int KDIM = FF + HH;     // 384
constexpr int ZSTR = KDIM + 8;    // 392: padded LDS row stride (bank-conflict-free b128 loads)
constexpr int ROWS = 32;          // batch rows per workgroup

#if __has_builtin(__builtin_amdgcn_sched_group_barrier)
#define SGB(mask, cnt, sync) __builtin_amdgcn_sched_group_barrier(mask, cnt, sync)
#else
#define SGB(mask, cnt, sync)
#endif

union AFrag { v16bf v; v8bf h[2]; };

__device__ __forceinline__ float sigm_(float x) { return 1.0f / (1.0f + __expf(-x)); }
__device__ __forceinline__ float tanh_(float x) { return 2.0f / (1.0f + __expf(-2.0f * x)) - 1.0f; }

// Pack 4 fp32 gate weight matrices [256,384] into contiguous bf16 [4][256][384].
__global__ __launch_bounds__(256)
void pack_weights(const float* __restrict__ Wf, const float* __restrict__ Wu,
                  const float* __restrict__ Wc, const float* __restrict__ Wo,
                  unsigned short* __restrict__ out) {
    const int per = HH * KDIM;                       // 98304
    int i = blockIdx.x * blockDim.x + threadIdx.x;
    if (i >= 4 * per) return;
    const float* src = (i < per) ? Wf : (i < 2 * per) ? Wu : (i < 3 * per) ? Wc : Wo;
    __bf16 v = (__bf16)src[i % per];
    out[i] = *reinterpret_cast<unsigned short*>(&v);
}

__global__ __launch_bounds__(256)
void lstm_persistent(const float* __restrict__ x,
                     const unsigned short* __restrict__ Wpk,
                     const float* __restrict__ bfv, const float* __restrict__ buv,
                     const float* __restrict__ bcv, const float* __restrict__ bov,
                     const float* __restrict__ W1, const float* __restrict__ b1,
                     const float* __restrict__ W2, const float* __restrict__ b2,
                     float* __restrict__ out) {
    // Double-buffered [x_t | h] tile: read Z[t&1], write h(t) into Z[(t+1)&1].
    __shared__ __bf16 Zb[2][ROWS * ZSTR];

    const __bf16* Wb = reinterpret_cast<const __bf16*>(Wpk);

    const int tid  = threadIdx.x;
    const int wave = tid >> 5;           // 0..7: owns hidden cols [32*wave, 32*wave+32)
    const int lane = tid & 31;
    const int l16  = lane & 15;
    const int half = lane >> 4;          // 0: lanes 0-15, 1: lanes 16-31
    const int gb   = blockIdx.x * ROWS;

    // zero the h region of the first read buffer (h0 = 0)
    for (int i = tid; i < ROWS * HH; i += 256) {
        int r = i >> 8, c = i & 255;
        Zb[0][r * ZSTR + FF + c] = (__bf16)0.0f;
    }

    // persistent cell state: [jj][mtile], C/D register layout
    v8f cst[2][2];
#pragma unroll
    for (int jj = 0; jj < 2; ++jj)
#pragma unroll
        for (int m = 0; m < 2; ++m)
#pragma unroll
            for (int i = 0; i < 8; ++i) cst[jj][m][i] = 0.0f;

    const int xr = tid >> 3;             // 0..31 : batch row for x loader
    const int xc = (tid & 7) * 16;       // 16 floats per thread

    const int koff = half * 8;           // A frag: lanes<16 K{0..7,16..23}, lanes>=16 K{8..15,24..31}
    const int kh   = half * 8;           // B frag: lanes<16 K{0..7,16..23}, lanes>=16 K{8..15,24..31}

    for (int t = 0; t < TT; ++t) {
        __bf16* Zr = Zb[t & 1];          // holds x_t (below) and h(t-1)
        __bf16* Zw = Zb[(t + 1) & 1];    // receives h(t)

        // Opaque zero: stops LICM from hoisting the (t-invariant) weight loads
        // out of the recurrence; weights re-stream from the 192MB L2 each step.
        int wzero;
        asm volatile("s_mov_b32 %0, 0" : "=s"(wzero));
        const __bf16* Wbt = Wb + wzero;

        // ---- stage x_t into Zr[:,0:128] as bf16 (fp32 HBM -> LDS) ----
        {
            const float4* xp = reinterpret_cast<const float4*>(
                x + ((size_t)(gb + xr) * TT + t) * FF + xc);
            __bf16* zr = &Zr[xr * ZSTR + xc];
#pragma unroll
            for (int i = 0; i < 4; ++i) {
                float4 v = xp[i];
                zr[4 * i + 0] = (__bf16)v.x; zr[4 * i + 1] = (__bf16)v.y;
                zr[4 * i + 2] = (__bf16)v.z; zr[4 * i + 3] = (__bf16)v.w;
            }
        }
        __syncthreads();   // x(t) writes + h(t-1) writes visible to all waves

        const __bf16* arow0 = &Zr[(0 * 16 + l16) * ZSTR + koff];   // M-tile 0
        const __bf16* arow1 = &Zr[(1 * 16 + l16) * ZSTR + koff];   // M-tile 1

#pragma unroll
        for (int jj = 0; jj < 2; ++jj) {
            const int ncol = wave * 32 + jj * 16 + l16;   // hidden column this lane owns
            const __bf16* wbase = Wbt + (size_t)ncol * KDIM + kh;

            // ---- software-pipelined K loop: double-buffered B fragments ----
            // Each B fragment (one L2 read) feeds TWO WMMAs (both M-tiles).
            AFrag bb[2][4];
#pragma unroll
            for (int gg = 0; gg < 4; ++gg) {
                const __bf16* wrow = wbase + (size_t)gg * (HH * KDIM);
                bb[0][gg].h[0] = *reinterpret_cast<const v8bf*>(wrow);
                bb[0][gg].h[1] = *reinterpret_cast<const v8bf*>(wrow + 16);
            }

            v8f acc[4][2];
#pragma unroll
            for (int gg = 0; gg < 4; ++gg)
#pragma unroll
                for (int m = 0; m < 2; ++m)
#pragma unroll
                    for (int i = 0; i < 8; ++i) acc[gg][m][i] = 0.0f;

#pragma unroll
            for (int k = 0; k < 12; ++k) {
                const int cur = k & 1, nxt = cur ^ 1;
                if (k < 11) {            // prefetch B(k+1) while k computes
#pragma unroll
                    for (int gg = 0; gg < 4; ++gg) {
                        const __bf16* wrow = wbase + (size_t)gg * (HH * KDIM) + (k + 1) * 32;
                        bb[nxt][gg].h[0] = *reinterpret_cast<const v8bf*>(wrow);
                        bb[nxt][gg].h[1] = *reinterpret_cast<const v8bf*>(wrow + 16);
                    }
                }
                // A fragments for both M-tiles (cheap LDS reads, loaded at use)
                AFrag a0, a1;
                a0.h[0] = *reinterpret_cast<const v8bf*>(arow0 + k * 32);
                a0.h[1] = *reinterpret_cast<const v8bf*>(arow0 + k * 32 + 16);
                a1.h[0] = *reinterpret_cast<const v8bf*>(arow1 + k * 32);
                a1.h[1] = *reinterpret_cast<const v8bf*>(arow1 + k * 32 + 16);

#pragma unroll
                for (int gg = 0; gg < 4; ++gg) {
                    acc[gg][0] = __builtin_amdgcn_wmma_f32_16x16x32_bf16(
                        false, a0.v, false, bb[cur][gg].v, (short)0, acc[gg][0], false, false);
                    acc[gg][1] = __builtin_amdgcn_wmma_f32_16x16x32_bf16(
                        false, a1.v, false, bb[cur][gg].v, (short)0, acc[gg][1], false, false);
                }
                // Enforce pipeline order per k-step: issue the 8 prefetch loads,
                // then the 4 LDS A-loads, then the 8 WMMAs.
                SGB(0x020, 8, 0);   // VMEM read  x8 (B prefetch)
                SGB(0x100, 4, 0);   // DS   read  x4 (A frags)
                SGB(0x008, 8, 0);   // MFMA/WMMA  x8
            }

            // ---- add biases, fused cell update in the C/D register layout ----
            const float c0 = bfv[ncol], c1 = buv[ncol], c2 = bcv[ncol], c3 = bov[ncol];
#pragma unroll
            for (int m = 0; m < 2; ++m) {
                v8f hnew;
#pragma unroll
                for (int i = 0; i < 8; ++i) {
                    float f  = sigm_(acc[0][m][i] + c0);
                    float u  = sigm_(acc[1][m][i] + c1);
                    float gv = tanh_(acc[2][m][i] + c2);
                    float o  = sigm_(acc[3][m][i] + c3);
                    float c  = f * cst[jj][m][i] + u * gv;
                    cst[jj][m][i] = c;
                    hnew[i] = o * tanh_(c);
                }
#pragma unroll
                for (int i = 0; i < 8; ++i) {
                    int row = m * 16 + i + half * 8;
                    Zw[row * ZSTR + FF + ncol] = (__bf16)hnew[i];
                }
            }
        }
    }
    __syncthreads();

    // ---- dense head: y = (h @ W1^T + b1) @ W2^T + b2 ----
    // TT even -> final h lives in Zb[0][:,128:384] (bf16).
    // All 256 threads: 8 threads per batch row, 8 of the 64 W1 rows each,
    // then a 3-step shfl_xor reduction across the 8 collaborating lanes.
    {
        const int r      = tid >> 3;         // batch row 0..31
        const int dchunk = tid & 7;          // which 8 output dims
        const __bf16* hrow = &Zb[0][r * ZSTR + FF];
        float part = 0.0f;
        for (int dd = 0; dd < 8; ++dd) {
            const int d = dchunk * 8 + dd;
            float s = b1[d];
            const float* w = W1 + d * HH;
            for (int k = 0; k < HH; ++k) s += (float)hrow[k] * w[k];
            part += W2[d] * s;
        }
        part += __shfl_xor(part, 1);
        part += __shfl_xor(part, 2);
        part += __shfl_xor(part, 4);
        if (dchunk == 0) out[gb + r] = part + b2[0];
    }
}

extern "C" void kernel_launch(void* const* d_in, const int* in_sizes, int n_in,
                              void* d_out, int out_size, void* d_ws, size_t ws_size,
                              hipStream_t stream) {
    (void)in_sizes; (void)n_in; (void)out_size; (void)ws_size;
    const float* x  = (const float*)d_in[0];
    const float* Wf = (const float*)d_in[1];
    const float* bf = (const float*)d_in[2];
    const float* Wu = (const float*)d_in[3];
    const float* bu = (const float*)d_in[4];
    const float* Wc = (const float*)d_in[5];
    const float* bc = (const float*)d_in[6];
    const float* Wo = (const float*)d_in[7];
    const float* bo = (const float*)d_in[8];
    const float* W1 = (const float*)d_in[9];
    const float* b1 = (const float*)d_in[10];
    const float* W2 = (const float*)d_in[11];
    const float* b2 = (const float*)d_in[12];

    unsigned short* Wpk = (unsigned short*)d_ws;   // 4*256*384*2 = 786432 bytes

    const int total = 4 * HH * KDIM;
    pack_weights<<<(total + 255) / 256, 256, 0, stream>>>(Wf, Wu, Wc, Wo, Wpk);
    lstm_persistent<<<BB / ROWS, 256, 0, stream>>>(
        x, Wpk, bf, bu, bc, bo, W1, b1, W2, b2, (float*)d_out);
}
